// GraphAttentionLayer_1726576853068
// MI455X (gfx1250) — matrix operand order repaired and checked by reading, not compiled, not run on hardware
//
#include <hip/hip_runtime.h>
#include <hip/hip_bf16.h>

typedef __attribute__((ext_vector_type(16))) __bf16 v16bf;
typedef __attribute__((ext_vector_type(8)))  __bf16 v8bf;
typedef __attribute__((ext_vector_type(4)))  __bf16 v4bf;
typedef __attribute__((ext_vector_type(8)))  float  v8f;

#define N_NODES  8192
#define FDIM     256
#define NE       262144
#define ALPHA_LR 0.2f

// ---------------------------------------------------------------------------
// Pre-conversion: h -> bf16 row-major; W -> bf16 TRANSPOSED (col-major),
// so both WMMA fragments load as contiguous b128s.
// ---------------------------------------------------------------------------
__global__ __launch_bounds__(256) void k_cvt_bf16(const float* __restrict__ x,
                                                  __bf16* __restrict__ y) {
    size_t i = ((size_t)blockIdx.x * blockDim.x + threadIdx.x) * 4;
    float4 v = *(const float4*)(x + i);
    v4bf o;
    o[0] = (__bf16)v.x; o[1] = (__bf16)v.y; o[2] = (__bf16)v.z; o[3] = (__bf16)v.w;
    *(v4bf*)(y + i) = o;
}

__global__ __launch_bounds__(256) void k_cvt_wt(const float* __restrict__ W,
                                                __bf16* __restrict__ Wt) {
    int k = threadIdx.x;   // 0..255
    int n = blockIdx.x;    // 0..255
    Wt[(size_t)n * FDIM + k] = (__bf16)W[(size_t)k * FDIM + n];
}

// ---------------------------------------------------------------------------
// Kernel 1: Wh = h @ W  (bf16 WMMA, f32 accumulate). One 16x16 tile per wave.
// A (16x32 bf16): lane M = lane&15; K pairs at half*8 + {0..7} and 16+half*8+{0..7}
// B (32x16 bf16): lane N = lane&15; K pairs at half*8 + {0..7} and 16+half*8+{0..7} (from W^T)
// ---------------------------------------------------------------------------
__global__ __launch_bounds__(256) void k_gemm_wh(const __bf16* __restrict__ hb,
                                                 const __bf16* __restrict__ Wtb,
                                                 float* __restrict__ Wh) {
    const int lane = threadIdx.x & 31;
    const int wave = threadIdx.x >> 5;
    const int tile = blockIdx.x * 8 + wave;   // 0..8191
    const int tm   = tile >> 4;               // row tile 0..511
    const int tn   = tile & 15;               // col tile 0..15
    const int half = lane >> 4;
    const int l15  = lane & 15;

    v8f c = {};
    const __bf16* arow = hb  + (size_t)(tm * 16 + l15) * FDIM;  // A: M = l15
    const __bf16* brow = Wtb + (size_t)(tn * 16 + l15) * FDIM;  // B: N = l15 (W^T)

    #pragma unroll
    for (int kk = 0; kk < FDIM; kk += 32) {
        v8bf alo = *(const v8bf*)(arow + kk + half * 8);        // K = half*8+0..7
        v8bf ahi = *(const v8bf*)(arow + kk + 16 + half * 8);   // K = 16+half*8+0..7
        v8bf blo = *(const v8bf*)(brow + kk + half * 8);        // K = half*8+0..7
        v8bf bhi = *(const v8bf*)(brow + kk + 16 + half * 8);   // K = 16+half*8+0..7
        v16bf b  = __builtin_shufflevector(blo, bhi, 0,1,2,3,4,5,6,7,8,9,10,11,12,13,14,15);
        v16bf a;
        #pragma unroll
        for (int j = 0; j < 8; ++j) { a[j] = alo[j]; a[8 + j] = ahi[j]; }
        c = __builtin_amdgcn_wmma_f32_16x16x32_bf16(false, a, false, b,
                                                    (short)0, c, false, false);
    }
    #pragma unroll
    for (int v = 0; v < 8; ++v) {
        int row = tm * 16 + half * 8 + v;                       // C/D VGPR layout
        Wh[(size_t)row * FDIM + tn * 16 + l15] = c[v];
    }
}

// ---------------------------------------------------------------------------
// Wave reduction (wave32)
// ---------------------------------------------------------------------------
__device__ inline float wave_reduce_sum(float v) {
    #pragma unroll
    for (int m = 16; m >= 1; m >>= 1) v += __shfl_xor(v, m, 32);
    return v;
}

// ---------------------------------------------------------------------------
// Kernel 2: s_src[i] = Wh[i,:]·a1 ; s_dst[i] = Wh[i,:]·a2  (one wave per row)
// ---------------------------------------------------------------------------
__global__ __launch_bounds__(256) void k_node_scores(const float* __restrict__ Wh,
                                                     const float* __restrict__ a,
                                                     float* __restrict__ s_src,
                                                     float* __restrict__ s_dst) {
    const int lane = threadIdx.x & 31;
    const int row  = blockIdx.x * 8 + (threadIdx.x >> 5);
    const float* r = Wh + (size_t)row * FDIM;
    float p1 = 0.f, p2 = 0.f;
    for (int j = lane; j < FDIM; j += 32) {
        float x = r[j];
        p1 += x * a[j];
        p2 += x * a[FDIM + j];
    }
    p1 = wave_reduce_sum(p1);
    p2 = wave_reduce_sum(p2);
    if (lane == 0) { s_src[row] = p1; s_dst[row] = p2; }
}

// ---------------------------------------------------------------------------
// Kernel 3: e_scores[e] = edge_attr[e,:]·a3  (one wave per edge)
// ---------------------------------------------------------------------------
__global__ __launch_bounds__(256) void k_edge_scores(const float* __restrict__ ea,
                                                     const float* __restrict__ a3,
                                                     float* __restrict__ esc) {
    const int lane = threadIdx.x & 31;
    const int e    = blockIdx.x * 8 + (threadIdx.x >> 5);
    const float* r = ea + (size_t)e * FDIM;
    float p = 0.f;
    for (int j = lane; j < FDIM; j += 32) p += r[j] * a3[j];
    p = wave_reduce_sum(p);
    if (lane == 0) esc[e] = p;
}

// ---------------------------------------------------------------------------
// CSR construction: counts (diag=1 + both edge directions), scan, fill.
// ---------------------------------------------------------------------------
__global__ void k_init_cnt(int* cnt) {
    int i = blockIdx.x * blockDim.x + threadIdx.x;
    if (i < N_NODES) cnt[i] = 1;            // diagonal entry
}

__global__ void k_count(const int* __restrict__ adj, int* __restrict__ cnt) {
    int e = blockIdx.x * blockDim.x + threadIdx.x;
    if (e >= NE) return;
    int s = adj[e], d = adj[NE + e];
    atomicAdd(&cnt[s], 1);                  // row s gets neighbor d
    atomicAdd(&cnt[d], 1);                  // row d gets neighbor s (mask symm.)
}

__global__ __launch_bounds__(256) void k_scan(const int* __restrict__ cnt,
                                              int* __restrict__ off,
                                              int* __restrict__ cursor) {
    __shared__ int part[256];
    const int t = threadIdx.x;
    const int base = t * 32;
    int local = 0;
    int c[32];
    #pragma unroll
    for (int j = 0; j < 32; ++j) { c[j] = cnt[base + j]; local += c[j]; }
    part[t] = local;
    __syncthreads();
    for (int o = 1; o < 256; o <<= 1) {     // Hillis-Steele inclusive scan
        int v = (t >= o) ? part[t - o] : 0;
        __syncthreads();
        part[t] += v;
        __syncthreads();
    }
    int run = part[t] - local;              // exclusive prefix for this chunk
    #pragma unroll
    for (int j = 0; j < 32; ++j) {
        off[base + j]    = run;
        cursor[base + j] = run;
        run += c[j];
    }
    if (t == 255) off[N_NODES] = part[255];
}

__global__ void k_fill_diag(int* __restrict__ cursor, int* __restrict__ col,
                            float* __restrict__ val) {
    int i = blockIdx.x * blockDim.x + threadIdx.x;
    if (i >= N_NODES) return;
    int p = atomicAdd(&cursor[i], 1);
    col[p] = i; val[p] = 0.f;
}

__global__ void k_fill_edges(const int* __restrict__ adj, const float* __restrict__ esc,
                             int* __restrict__ cursor, int* __restrict__ col,
                             float* __restrict__ val) {
    int e = blockIdx.x * blockDim.x + threadIdx.x;
    if (e >= NE) return;
    int s = adj[e], d = adj[NE + e];
    float sc = esc[e];
    int p = atomicAdd(&cursor[s], 1);       // forward edge: carries e_edge score
    col[p] = d; val[p] = sc;
    int q = atomicAdd(&cursor[d], 1);       // reverse direction: e_edge = 0
    col[q] = s; val[q] = 0.f;
}

// ---------------------------------------------------------------------------
// Kernel 4: fused masked softmax + aggregation + ELU. One block per row;
// thread t owns output feature t. Neighbor (col,val) chunks are staged into
// LDS with CDNA5 GLOBAL_LOAD_ASYNC_TO_LDS (ASYNCcnt), and the Wh rows about
// to be gathered are prefetched (global_prefetch_b8).
// ---------------------------------------------------------------------------
__global__ __launch_bounds__(256) void k_aggregate(const float* __restrict__ Wh,
                                                   const int* __restrict__ off,
                                                   const int* __restrict__ col,
                                                   const float* __restrict__ val,
                                                   const float* __restrict__ s_src,
                                                   const float* __restrict__ s_dst,
                                                   float* __restrict__ out) {
    __shared__ int   csh[256];   // async-staged neighbor columns
    __shared__ float vsh[256];   // async-staged edge scores
    __shared__ float wsh[256];   // softmax weights
    __shared__ float red[256];
    const int i  = blockIdx.x;
    const int t  = threadIdx.x;
    const int b0 = off[i], b1 = off[i + 1];
    const float ssi = s_src[i];

    // Pass 1: leaky-relu scores -> row max
    float mx = -3.0e38f;
    for (int n = b0 + t; n < b1; n += 256) {
        float e = ssi + s_dst[col[n]] + val[n];
        e = e > 0.f ? e : ALPHA_LR * e;
        mx = fmaxf(mx, e);
    }
    red[t] = mx;
    __syncthreads();
    for (int o = 128; o >= 1; o >>= 1) {
        if (t < o) red[t] = fmaxf(red[t], red[t + o]);
        __syncthreads();
    }
    mx = red[0];
    __syncthreads();

    // Pass 2: async-stage chunk into LDS, exp weights, per-feature accumulate
    float acc = 0.f, sumw = 0.f;
    for (int base = b0; base < b1; base += 256) {
        const int len = min(256, b1 - base);
        if (t < len) {
            unsigned lds_c = (unsigned)(size_t)&csh[t];
            unsigned lds_v = (unsigned)(size_t)&vsh[t];
            const int*   gc = col + base + t;
            const float* gv = val + base + t;
            asm volatile("global_load_async_to_lds_b32 %0, %1, off"
                         :: "v"(lds_c), "v"(gc) : "memory");
            asm volatile("global_load_async_to_lds_b32 %0, %1, off"
                         :: "v"(lds_v), "v"(gv) : "memory");
        }
        asm volatile("s_wait_asynccnt 0x0" ::: "memory");
        __syncthreads();
        if (t < len) {
            int cidx = csh[t];
            float e = ssi + s_dst[cidx] + vsh[t];
            e = e > 0.f ? e : ALPHA_LR * e;
            float w = __expf(e - mx);
            wsh[t] = w;
            sumw += w;
            __builtin_prefetch(&Wh[(size_t)cidx * FDIM], 0, 1);  // warm WGP$
        }
        __syncthreads();
        for (int n = 0; n < len; ++n)
            acc += wsh[n] * Wh[(size_t)csh[n] * FDIM + t];
        __syncthreads();
    }
    red[t] = sumw;
    __syncthreads();
    for (int o = 128; o >= 1; o >>= 1) {
        if (t < o) red[t] += red[t + o];
        __syncthreads();
    }
    sumw = red[0];

    float hp = acc / sumw;
    out[(size_t)i * FDIM + t] = hp > 0.f ? hp : (__expf(hp) - 1.f);
}

// ---------------------------------------------------------------------------
// Host launcher
// ---------------------------------------------------------------------------
extern "C" void kernel_launch(void* const* d_in, const int* in_sizes, int n_in,
                              void* d_out, int out_size, void* d_ws, size_t ws_size,
                              hipStream_t stream) {
    const float* h   = (const float*)d_in[0];   // (8192, 256)
    const int*   adj = (const int*)  d_in[1];   // (2, 262144)
    const float* ea  = (const float*)d_in[2];   // (262144, 256)
    const float* W   = (const float*)d_in[3];   // (256, 256)
    const float* a   = (const float*)d_in[4];   // (768, 1)
    float* out = (float*)d_out;

    // bump allocator over d_ws (~18 MB total)
    char* ws = (char*)d_ws;
    size_t o = 0;
    auto alloc = [&](size_t bytes) -> void* {
        o = (o + 255) & ~(size_t)255;
        void* p = ws + o;
        o += bytes;
        return p;
    };
    float*  Wh     = (float*) alloc((size_t)N_NODES * FDIM * 4);   // 8 MB
    __bf16* hb     = (__bf16*)alloc((size_t)N_NODES * FDIM * 2);   // 4 MB
    __bf16* Wtb    = (__bf16*)alloc((size_t)FDIM * FDIM * 2);      // 128 KB
    float*  ssrc   = (float*) alloc(N_NODES * 4);
    float*  sdst   = (float*) alloc(N_NODES * 4);
    float*  esc    = (float*) alloc((size_t)NE * 4);               // 1 MB
    int*    cnt    = (int*)   alloc(N_NODES * 4);
    int*    off    = (int*)   alloc((N_NODES + 1) * 4);
    int*    cursor = (int*)   alloc(N_NODES * 4);
    const size_t cap = (size_t)N_NODES + 2 * (size_t)NE;           // 532480
    int*    col    = (int*)   alloc(cap * 4);                      // ~2.1 MB
    float*  val    = (float*) alloc(cap * 4);                      // ~2.1 MB
    (void)ws_size; (void)in_sizes; (void)n_in; (void)out_size;

    k_cvt_bf16  <<<(N_NODES * FDIM / 4) / 256, 256, 0, stream>>>(h, hb);
    k_cvt_wt    <<<FDIM, FDIM, 0, stream>>>(W, Wtb);
    k_gemm_wh   <<<N_NODES * (FDIM / 16) / (16 * 8), 256, 0, stream>>>(hb, Wtb, Wh);
    k_node_scores<<<N_NODES / 8, 256, 0, stream>>>(Wh, a, ssrc, sdst);
    k_edge_scores<<<NE / 8, 256, 0, stream>>>(ea, a + 2 * FDIM, esc);
    k_init_cnt  <<<N_NODES / 256, 256, 0, stream>>>(cnt);
    k_count     <<<NE / 256, 256, 0, stream>>>(adj, cnt);
    k_scan      <<<1, 256, 0, stream>>>(cnt, off, cursor);
    k_fill_diag <<<N_NODES / 256, 256, 0, stream>>>(cursor, col, val);
    k_fill_edges<<<NE / 256, 256, 0, stream>>>(adj, esc, cursor, col, val);
    k_aggregate <<<N_NODES, 256, 0, stream>>>(Wh, off, col, val, ssrc, sdst, out);
}
